// DistLayer_88794153877511
// MI455X (gfx1250) — compile-verified, matrix operand order checked by reading, not run on hardware
//
#include <hip/hip_runtime.h>
#include <hip/hip_bf16.h>
#include <cstdint>

#define N_ROWS 131072
#define NSEG   4096
#define KTOT   640
#define NOUT   256
#define EPSV   1e-5f

typedef __attribute__((ext_vector_type(16))) __bf16       v16bf;
typedef __attribute__((ext_vector_type(8)))  float        v8f;
typedef __attribute__((ext_vector_type(4)))  unsigned int u32x4;
typedef __attribute__((ext_vector_type(8)))  int          i32x8;
typedef __attribute__((ext_vector_type(4)))  int          i32x4;

// ---------------------------------------------------------------- zero ws
__global__ __launch_bounds__(256) void zero_ws(float* __restrict__ p, int n) {
    int i = blockIdx.x * 256 + threadIdx.x;
    if (i < n) p[i] = 0.0f;
}

// ------------------------------------------------- segment sums (atomics)
__global__ __launch_bounds__(128) void seg_accum(const float* __restrict__ x,
                                                 const int* __restrict__ aidx,
                                                 float* __restrict__ sumAtom,
                                                 float* __restrict__ sumEle,
                                                 float* __restrict__ cnt) {
    const int row = blockIdx.x;
    const int t   = threadIdx.x;
    const int seg = aidx[row];
    const float* xr = x + (size_t)row * 384;
    atomicAdd(&sumAtom[(size_t)seg * 128 + t], xr[t]);
    atomicAdd(&sumEle [(size_t)seg * 128 + t], xr[128 + t]);
    if (t == 0) atomicAdd(&cnt[seg], 1.0f);
}

// ------------------------------------------------------ means -> bf16
__global__ __launch_bounds__(128) void seg_means(const float* __restrict__ sumAtom,
                                                 const float* __restrict__ sumEle,
                                                 const float* __restrict__ cnt,
                                                 __bf16* __restrict__ mA,
                                                 __bf16* __restrict__ mE) {
    const int s = blockIdx.x;
    const int t = threadIdx.x;
    const float inv = 1.0f / fmaxf(cnt[s], 1.0f);
    mA[(size_t)s * 128 + t] = (__bf16)(sumAtom[(size_t)s * 128 + t] * inv);
    mE[(size_t)s * 128 + t] = (__bf16)(sumEle [(size_t)s * 128 + t] * inv);
}

// ---------------- pack W1 (640x256 f32 row-major) into WMMA B fragments
// layout: [ktile 20][ntile 16][lane 32][16 bf16] -> each 16KB kt-slice is
// contiguous so the TDM can stream one slice per K step.
// lane n<16 holds col n, K = kt*32 + (lane>>4)*16 + j  (contiguous k order, A uses the same)
__global__ __launch_bounds__(256) void pack_w(const float* __restrict__ W1,
                                              __bf16* __restrict__ wpack) {
    const int tid = blockIdx.x * 256 + threadIdx.x;   // 0 .. 10239
    if (tid >= 20 * 16 * 32) return;
    const int kt   = tid / 512;
    const int rem  = tid % 512;
    const int nt   = rem / 32;
    const int lane = rem % 32;
    const int n  = nt * 16 + (lane & 15);
    const int kb = kt * 32 + (lane >> 4) * 16;
    v16bf v;
#pragma unroll
    for (int j = 0; j < 16; ++j) v[j] = (__bf16)W1[(size_t)(kb + j) * NOUT + n];
    *(v16bf*)(wpack + (size_t)tid * 16) = v;
}

// ---------------------------------------------------------------- GEMM
__device__ inline v16bf cvt16(const float* __restrict__ p) {
    const float4* p4 = (const float4*)p;
    float4 f0 = p4[0], f1 = p4[1], f2 = p4[2], f3 = p4[3];
    v16bf a;
    a[0]=(__bf16)f0.x; a[1]=(__bf16)f0.y; a[2]=(__bf16)f0.z; a[3]=(__bf16)f0.w;
    a[4]=(__bf16)f1.x; a[5]=(__bf16)f1.y; a[6]=(__bf16)f1.z; a[7]=(__bf16)f1.w;
    a[8]=(__bf16)f2.x; a[9]=(__bf16)f2.y; a[10]=(__bf16)f2.z; a[11]=(__bf16)f2.w;
    a[12]=(__bf16)f3.x; a[13]=(__bf16)f3.y; a[14]=(__bf16)f3.z; a[15]=(__bf16)f3.w;
    return a;
}

// A row = [x_atom | pooled_atom | x_ele | pooled_ele | x_dist], assembled on the fly
__device__ inline v16bf load_a_frag(const float* __restrict__ xrow,
                                    const __bf16* __restrict__ mAt,
                                    const __bf16* __restrict__ mEl,
                                    int kb) {
    if (kb < 128)       return cvt16(xrow + kb);                    // x_atom
    else if (kb < 256)  return *(const v16bf*)(mAt + (kb - 128));   // pooled_atom
    else if (kb < 384)  return cvt16(xrow + (kb - 128));            // x_ele
    else if (kb < 512)  return *(const v16bf*)(mEl + (kb - 384));   // pooled_ele
    else                return cvt16(xrow + (kb - 256));            // x_dist
}

// issue one TDM load of a contiguous 16KB B-slice into LDS (offset ldsOff)
__device__ inline void tdm_load_slice(const __bf16* __restrict__ gsrc,
                                      unsigned ldsOff) {
#if __has_builtin(__builtin_amdgcn_tensor_load_to_lds)
    unsigned long long gaddr =
        (unsigned long long)reinterpret_cast<uintptr_t>(gsrc);
    u32x4 g0;
    g0[0] = 1u;                                          // count=1, user mode
    g0[1] = ldsOff;                                      // lds_addr
    g0[2] = (unsigned)(gaddr & 0xffffffffull);           // global_addr[31:0]
    g0[3] = (unsigned)((gaddr >> 32) & 0x1ffffffull)     // global_addr[56:32]
          | (2u << 30);                                  // type = 2 ("image")
    i32x8 g1;
    g1[0] = 0x00030000;          // data_size = 8B, no multicast/iterate/pad
    g1[1] = (int)(2048u << 16);  // tensor_dim0 = 2048 elements (16384 B)
    g1[2] = 0x00010000;          // tensor_dim0[31:16]=0, tensor_dim1 = 1
    g1[3] = (int)(2048u << 16);  // tile_dim0 = 2048
    g1[4] = 1;                   // tile_dim1 = 1
    g1[5] = 2048;                // tensor_dim0_stride
    g1[6] = 0;
    g1[7] = 0;
    i32x4 gz4 = {0, 0, 0, 0};
    i32x8 gz8 = {0, 0, 0, 0, 0, 0, 0, 0};
    __builtin_amdgcn_tensor_load_to_lds(g0, g1, gz4, gz4, gz8, 0);
#endif
}

__global__ __launch_bounds__(256) void gemm_wmma(const float* __restrict__ x,
                                                 const int* __restrict__ aidx,
                                                 const int* __restrict__ eidx,
                                                 const __bf16* __restrict__ meanAtom,
                                                 const __bf16* __restrict__ meanEle,
                                                 const __bf16* __restrict__ wpack,
                                                 const float* __restrict__ b1,
                                                 float* __restrict__ hout) {
    extern __shared__ __align__(16) char ldsRaw[];   // 2 x 16384 B ring of B slices
    __bf16* ldsB = (__bf16*)ldsRaw;

    const int wave = threadIdx.x >> 5;
    const int lane = threadIdx.x & 31;
    const int rowBlock = blockIdx.x * 128;

    // each wave owns one 16-row tile and all 16 col-tiles (full N=256):
    // x is read from HBM exactly once per row.
    const int r = rowBlock + wave * 16 + (lane & 15);
    const int khalf = (lane >> 4) * 16;

    const float*  xr = x + (size_t)r * 384;
    const __bf16* mAr = meanAtom + (size_t)aidx[r] * 128;
    const __bf16* mEr = meanEle  + (size_t)eidx[r] * 128;

    // prologue: stream kt=0 B slice via Tensor Data Mover
#if __has_builtin(__builtin_amdgcn_tensor_load_to_lds)
    if (threadIdx.x < 32) {
        tdm_load_slice(wpack, 0u);
        __builtin_amdgcn_s_wait_tensorcnt(0);
    }
    __syncthreads();
#else
    {
        const ulonglong2* s = (const ulonglong2*)wpack;
        ulonglong2* d = (ulonglong2*)ldsB;
        for (int i = threadIdx.x; i < 16384 / 16; i += 256) d[i] = s[i];
    }
    __syncthreads();
#endif

    v8f acc[16] = {};

    for (int kt = 0; kt < 20; ++kt) {
        // kick off next slice into the other ring buffer while we compute
#if __has_builtin(__builtin_amdgcn_tensor_load_to_lds)
        if (kt + 1 < 20 && threadIdx.x < 32)
            tdm_load_slice(wpack + (size_t)(kt + 1) * 8192,
                           (unsigned)(((kt + 1) & 1) * 16384));
#else
        if (kt + 1 < 20) {
            const ulonglong2* s =
                (const ulonglong2*)(wpack + (size_t)(kt + 1) * 8192);
            ulonglong2* d =
                (ulonglong2*)(ldsRaw + ((kt + 1) & 1) * 16384);
            for (int i = threadIdx.x; i < 16384 / 16; i += 256) d[i] = s[i];
        }
#endif
        const __bf16* buf = (const __bf16*)(ldsRaw + (kt & 1) * 16384);
        const int kb = kt * 32 + khalf;
        v16bf a = load_a_frag(xr, mAr, mEr, kb);
#pragma unroll
        for (int nt = 0; nt < 16; ++nt) {
            v16bf b = *(const v16bf*)(buf + (size_t)(nt * 32 + lane) * 16);
            acc[nt] = __builtin_amdgcn_wmma_f32_16x16x32_bf16(
                false, a, false, b, (short)0, acc[nt], false, false);
        }
#if __has_builtin(__builtin_amdgcn_tensor_load_to_lds)
        if (kt + 1 < 20 && threadIdx.x < 32)
            __builtin_amdgcn_s_wait_tensorcnt(0);
#endif
        __syncthreads();
    }

    // epilogue: C/D layout (ISA 7.12.2): lane<16 -> M=v, lane>=16 -> M=v+8
    const int moff = (lane < 16) ? 0 : 8;
    const int rbase = rowBlock + wave * 16 + moff;
#pragma unroll
    for (int nt = 0; nt < 16; ++nt) {
        const int n = nt * 16 + (lane & 15);
        const float bias = b1[n];
#pragma unroll
        for (int v = 0; v < 8; ++v)
            hout[(size_t)(rbase + v) * NOUT + n] = acc[nt][v] + bias;
    }
}

// --------------------------------------------- column stats (mean / var)
__global__ __launch_bounds__(256) void col_stats(const float* __restrict__ h,
                                                 float* __restrict__ colSum,
                                                 float* __restrict__ colSq) {
    const int col = threadIdx.x;
    const int rowsPerBlock = N_ROWS / gridDim.x;
    const int r0 = blockIdx.x * rowsPerBlock;
    float s = 0.0f, q = 0.0f;
    for (int r = 0; r < rowsPerBlock; ++r) {
        float v = h[(size_t)(r0 + r) * NOUT + col];
        s += v;
        q += v * v;
    }
    atomicAdd(&colSum[col], s);
    atomicAdd(&colSq[col], q);
}

__global__ void finalize_stats(const float* __restrict__ colSum,
                               const float* __restrict__ colSq,
                               const float* __restrict__ gamma,
                               const float* __restrict__ beta,
                               float* __restrict__ scale,
                               float* __restrict__ shift) {
    const int t = threadIdx.x;
    const float inv = 1.0f / (float)N_ROWS;
    float mu  = colSum[t] * inv;
    float var = colSq[t] * inv - mu * mu;
    float rs  = rsqrtf(var + EPSV);
    float sc  = gamma[t] * rs;
    scale[t] = sc;
    shift[t] = beta[t] - mu * sc;
}

// ----------------------------------------------- BN + ReLU, in place on h
__global__ __launch_bounds__(256) void bn_relu(float* __restrict__ h,
                                               const float* __restrict__ scale,
                                               const float* __restrict__ shift) {
    __shared__ float sSc[256], sSh[256];
    sSc[threadIdx.x] = scale[threadIdx.x];
    sSh[threadIdx.x] = shift[threadIdx.x];
    __syncthreads();
    const size_t total4 = (size_t)N_ROWS * NOUT / 4;
    for (size_t i = (size_t)blockIdx.x * 256 + threadIdx.x; i < total4;
         i += (size_t)gridDim.x * 256) {
        float4 v = ((float4*)h)[i];
        const int n = (int)((i * 4) & 255);
        v.x = fmaxf(0.0f, v.x * sSc[n + 0] + sSh[n + 0]);
        v.y = fmaxf(0.0f, v.y * sSc[n + 1] + sSh[n + 1]);
        v.z = fmaxf(0.0f, v.z * sSc[n + 2] + sSh[n + 2]);
        v.w = fmaxf(0.0f, v.w * sSc[n + 3] + sSh[n + 3]);
        ((float4*)h)[i] = v;
    }
}

// ---------------------------------------------------------------- launch
extern "C" void kernel_launch(void* const* d_in, const int* in_sizes, int n_in,
                              void* d_out, int out_size, void* d_ws, size_t ws_size,
                              hipStream_t stream) {
    const float* x     = (const float*)d_in[0];
    const int*   aidx  = (const int*)  d_in[1];
    const int*   eidx  = (const int*)  d_in[2];
    const float* W1    = (const float*)d_in[3];
    const float* b1    = (const float*)d_in[4];
    const float* gamma = (const float*)d_in[5];
    const float* beta  = (const float*)d_in[6];
    float* h  = (float*)d_out;          // h lives in d_out (exactly N x 256 f32)
    float* ws = (float*)d_ws;

    // ws layout (float offsets); total ~6.4 MiB
    float*  sumAtom = ws;                    // 4096*128
    float*  sumEle  = ws + 524288;           // 4096*128
    float*  cnt     = ws + 1048576;          // 4096
    float*  colSum  = ws + 1052672;          // 256
    float*  colSq   = ws + 1052928;          // 256
    float*  scale   = ws + 1053184;          // 256
    float*  shift   = ws + 1053440;          // 256
    __bf16* mA      = (__bf16*)(ws + 1053696); // 4096*128 bf16
    __bf16* mE      = (__bf16*)(ws + 1315840); // 4096*128 bf16
    __bf16* wpack   = (__bf16*)(ws + 1577984); // 640*256 bf16 (kt-sliced fragments)

    const int nzero = 1053696;
    zero_ws<<<(nzero + 255) / 256, 256, 0, stream>>>(ws, nzero);
    seg_accum<<<N_ROWS, 128, 0, stream>>>(x, aidx, sumAtom, sumEle, cnt);
    seg_means<<<NSEG, 128, 0, stream>>>(sumAtom, sumEle, cnt, mA, mE);
    pack_w<<<40, 256, 0, stream>>>(W1, wpack);
    gemm_wmma<<<dim3(N_ROWS / 128), 256, 32768, stream>>>(
        x, aidx, eidx, mA, mE, wpack, b1, h);
    col_stats<<<256, 256, 0, stream>>>(h, colSum, colSq);
    finalize_stats<<<1, 256, 0, stream>>>(colSum, colSq, gamma, beta, scale, shift);
    bn_relu<<<4096, 256, 0, stream>>>(h, scale, shift);
}
